// CohomologicalTransformer_68212670595419
// MI455X (gfx1250) — matrix operand order repaired and checked by reading, not compiled, not run on hardware
//
#include <hip/hip_runtime.h>
#include <math.h>

// ---------------- model constants ----------------
#define DIMC   512
#define SEQD   64
#define VOCABC 30000
#define BATCH  2
#define SEQLEN 1024
#define ROWS   (BATCH * SEQLEN)   // 2048

typedef __attribute__((ext_vector_type(16))) __bf16 v16bf;
typedef __attribute__((ext_vector_type(8)))  float  v8f;

union Frag16 { v16bf v; uint4 u[2]; };
struct Frags { Frag16 a0, a1, b0, b1; };

static inline __device__ v8f wmma_bf16(const Frag16& a, const Frag16& b, v8f c) {
    return __builtin_amdgcn_wmma_f32_16x16x32_bf16(
        false, a.v, false, b.v, (short)0, c, false, false);
}

// ============================================================
// GEMM: C[M,N] = A[M,K](bf16) * Bt[Npad,K]^T(bf16) + bias
// Bt is zero-padded to Npad (multiple of 128) -> branchless main loop,
// EXEC stays all-1s through every WMMA. Tail masked only at store time.
// Wave tile 32(M) x 32(N), 2x2 WMMA accumulators, double-buffered frags.
// WG = 8 waves as 2(M) x 4(N)  ->  64 x 128 workgroup tile.
// MODE 0: plain, MODE 1: sigmoid
// ============================================================
template <int MODE>
__global__ __launch_bounds__(256) void gemm_bf16_kernel(
    const __bf16* __restrict__ A, const __bf16* __restrict__ Bt,
    const float* __restrict__ bias, const float* __restrict__ scale,
    float* __restrict__ Cf, __bf16* __restrict__ Cb,
    int M, int N, int K)
{
    const int wave = threadIdx.x >> 5;
    const int lane = threadIdx.x & 31;
    const int half = lane >> 4;     // 0 or 1
    const int l16  = lane & 15;

    const int wm = wave >> 2;       // 2 waves along M
    const int wn = wave & 3;        // 4 waves along N
    const int m0 = blockIdx.y * 64  + wm * 32;
    const int n0 = blockIdx.x * 128 + wn * 32;

    // A fragment (16x32 bf16): lanes 0-15/16-31 both carry M=l16;
    //   elems 0..7  = K + 8*half + (0..7), elems 8..15 = K + 16 + 8*half + (0..7)
    const __bf16* Ap0 = A + (size_t)(m0 + l16) * K + 8 * half;
    const __bf16* Ap1 = Ap0 + (size_t)16 * K;
    // B fragment (32x16 bf16) from Bt[Npad,K]: lane carries column n=l16;
    //   elems 0..7  = K + 8*half + (0..7), elems 8..15 = K + 16 + 8*half + (0..7)
    const __bf16* Bp0 = Bt + (size_t)(n0 + l16) * K + 8 * half;
    const __bf16* Bp1 = Bp0 + (size_t)16 * K;

    v8f acc00 = {}, acc01 = {}, acc10 = {}, acc11 = {};

    auto loadf = [&](Frags& f, int kb) {
        f.a0.u[0] = *(const uint4*)(Ap0 + kb);
        f.a0.u[1] = *(const uint4*)(Ap0 + kb + 16);
        f.a1.u[0] = *(const uint4*)(Ap1 + kb);
        f.a1.u[1] = *(const uint4*)(Ap1 + kb + 16);
        f.b0.u[0] = *(const uint4*)(Bp0 + kb);
        f.b0.u[1] = *(const uint4*)(Bp0 + kb + 16);
        f.b1.u[0] = *(const uint4*)(Bp1 + kb);
        f.b1.u[1] = *(const uint4*)(Bp1 + kb + 16);
    };
    auto mm = [&](const Frags& f) {
        acc00 = wmma_bf16(f.a0, f.b0, acc00);
        acc01 = wmma_bf16(f.a0, f.b1, acc01);
        acc10 = wmma_bf16(f.a1, f.b0, acc10);
        acc11 = wmma_bf16(f.a1, f.b1, acc11);
    };

    Frags f0, f1;
    loadf(f0, 0);
    for (int kb = 0; kb < K; kb += 64) {
        loadf(f1, kb + 32);           // prefetch while mm(f0) runs
        mm(f0);
        if (kb + 64 < K) loadf(f0, kb + 64);
        mm(f1);
    }

    // C/D layout: VGPR r -> row subtile + r + 8*half, column subtile + l16
    const v8f* accs[2][2] = { { &acc00, &acc01 }, { &acc10, &acc11 } };
    #pragma unroll
    for (int mi = 0; mi < 2; ++mi) {
        #pragma unroll
        for (int ni = 0; ni < 2; ++ni) {
            const int nc = n0 + ni * 16 + l16;
            if (nc < N) {
                const float bv = bias  ? bias[nc]  : 0.0f;
                const float sv = scale ? scale[nc] : 1.0f;
                #pragma unroll
                for (int r = 0; r < 8; ++r) {
                    const int mr = m0 + mi * 16 + r + 8 * half;
                    float v = (*accs[mi][ni])[r] + bv;
                    v *= sv;
                    if (MODE == 1) v = 1.0f / (1.0f + expf(-v));
                    const size_t idx = (size_t)mr * N + nc;
                    if (Cf) Cf[idx] = v;
                    if (Cb) Cb[idx] = (__bf16)v;
                }
            }
        }
    }
}

// ============================================================
// Windowed local attention. One block per (b, head, n).
// qkv: fp32 [ROWS, 3*DIMC] (q | k | v, each head-major h*hd+d)
// out: bf16 [ROWS, DIMC]
// ============================================================
__global__ __launch_bounds__(256) void attn_kernel(
    const float* __restrict__ qkv, __bf16* __restrict__ out,
    int heads, int hd, int window, float inv_sqrt_hd)
{
    __shared__ float q_s[256];
    __shared__ float sc[40];

    const int bid = blockIdx.x;
    const int h = bid % heads;
    const int n = (bid / heads) % SEQLEN;
    const int b = bid / (heads * SEQLEN);
    const int t = threadIdx.x;
    const int ld = 3 * DIMC;
    const int W = 2 * window + 1;

    if (t < hd)
        q_s[t] = qkv[(size_t)(b * SEQLEN + n) * ld + h * hd + t];
    __syncthreads();

    if (t < W) {
        const int m = n - window + t;
        float s = -1e30f;
        if (m >= 0 && m < SEQLEN) {
            const float* kp = qkv + (size_t)(b * SEQLEN + m) * ld + DIMC + h * hd;
            float acc = 0.0f;
            for (int d = 0; d < hd; ++d) acc += q_s[d] * kp[d];
            s = acc * inv_sqrt_hd;
        }
        sc[t] = s;
    }
    __syncthreads();

    if (t == 0) {
        float mx = -1e30f;
        for (int i = 0; i < W; ++i) mx = fmaxf(mx, sc[i]);
        float sum = 0.0f;
        for (int i = 0; i < W; ++i) { float e = expf(sc[i] - mx); sc[i] = e; sum += e; }
        const float inv = 1.0f / sum;
        for (int i = 0; i < W; ++i) sc[i] *= inv;
    }
    __syncthreads();

    if (t < hd) {
        float acc = 0.0f;
        for (int i = 0; i < W; ++i) {
            const int m = n - window + i;
            if (m >= 0 && m < SEQLEN)
                acc += sc[i] * qkv[(size_t)(b * SEQLEN + m) * ld + 2 * DIMC + h * hd + t];
        }
        out[(size_t)(b * SEQLEN + n) * DIMC + h * hd + t] = (__bf16)acc;
    }
}

// ============================================================
// Embedding gather -> bf16 activations
// ============================================================
__global__ void embed_kernel(const int* __restrict__ x,
                             const float* __restrict__ emb,
                             __bf16* __restrict__ hb)
{
    const int idx = blockIdx.x * blockDim.x + threadIdx.x;
    if (idx >= ROWS * DIMC) return;
    const int r = idx >> 9;          // / DIMC
    const int c = idx & (DIMC - 1);
    const int tok = x[r];
    hb[idx] = (__bf16)emb[(size_t)tok * DIMC + c];
}

// ============================================================
// Weight convert+transpose with zero padding:
// W[K,N] f32 -> Wt[Npad,K] bf16, rows n>=N are zero.
// Thread idx walks Wt (coalesced writes).
// ============================================================
__global__ void convtrans_kernel(const float* __restrict__ W,
                                 __bf16* __restrict__ Wt,
                                 int K, int N, int Npad)
{
    const int idx = blockIdx.x * blockDim.x + threadIdx.x;
    if (idx >= Npad * K) return;
    const int n = idx / K;
    const int k = idx - n * K;
    Wt[idx] = (n < N) ? (__bf16)W[(size_t)k * N + n] : (__bf16)0.0f;
}

// ============================================================
// Transition fuse: h = [seq*gate*emask , main] -> bf16
// ============================================================
__global__ void transition_fuse_kernel(
    const float* __restrict__ seqp, const float* __restrict__ gate,
    const float* __restrict__ mainp, const float* __restrict__ emask,
    __bf16* __restrict__ hb)
{
    const int idx = blockIdx.x * blockDim.x + threadIdx.x;
    if (idx >= ROWS * DIMC) return;
    const int r = idx >> 9;
    const int c = idx & (DIMC - 1);
    float v;
    if (c < SEQD)
        v = seqp[r * SEQD + c] * gate[r * SEQD + c] * emask[c];
    else
        v = mainp[r * (DIMC - SEQD) + (c - SEQD)];
    hb[idx] = (__bf16)v;
}

// ============================================================
// Host launcher
// ============================================================
static inline int npad128(int n) { return (n + 127) & ~127; }

extern "C" void kernel_launch(void* const* d_in, const int* in_sizes, int n_in,
                              void* d_out, int out_size, void* d_ws, size_t ws_size,
                              hipStream_t stream)
{
    (void)n_in; (void)out_size; (void)ws_size;

    const float *Wqkv[7], *bqkv[7], *Wp[7], *bp[7];
    const float *Wsp[2], *bsp[2], *Wmp[2], *bmp[2], *Wg[2], *bg[2], *emask[2];
    const float *Ws[3], *bs[3], *maskp[3];
    const float *Wout = nullptr, *bout = nullptr, *emb = nullptr;
    const int *x = nullptr;

    int i = 0;
    auto rd = [&]() { return (const float*)d_in[i++]; };

    if (in_sizes[0] == ROWS) {
        // setup_inputs() insertion order: x, emb, h0[...], t01, h1[...], t12, h2, tr0..2, Wout, bout
        x   = (const int*)d_in[i++];
        emb = rd();
        for (int l = 0; l < 4; ++l) { Wqkv[l]=rd(); bqkv[l]=rd(); Wp[l]=rd(); bp[l]=rd(); }
        Wsp[0]=rd(); bsp[0]=rd(); Wmp[0]=rd(); bmp[0]=rd(); Wg[0]=rd(); bg[0]=rd(); emask[0]=rd();
        for (int l = 4; l < 6; ++l) { Wqkv[l]=rd(); bqkv[l]=rd(); Wp[l]=rd(); bp[l]=rd(); }
        Wsp[1]=rd(); bsp[1]=rd(); Wmp[1]=rd(); bmp[1]=rd(); Wg[1]=rd(); bg[1]=rd(); emask[1]=rd();
        { const int l = 6; Wqkv[l]=rd(); bqkv[l]=rd(); Wp[l]=rd(); bp[l]=rd(); }
        for (int t = 0; t < 3; ++t) { Ws[t]=rd(); bs[t]=rd(); rd(); rd(); maskp[t]=rd(); }
        Wout = rd(); bout = rd();
    } else {
        // jax-style sorted-key flatten: Wout,bout,emb,h0..h2 (Wp,Wqkv,bp,bqkv),
        // t01,t12 (Wg,Wmp,Wsp,bg,bmp,bsp,emask), tr (Wo,Ws,bo,bs,mask), x
        Wout = rd(); bout = rd(); emb = rd();
        for (int l = 0; l < 7; ++l) { Wp[l]=rd(); Wqkv[l]=rd(); bp[l]=rd(); bqkv[l]=rd(); }
        for (int t = 0; t < 2; ++t) { Wg[t]=rd(); Wmp[t]=rd(); Wsp[t]=rd();
                                      bg[t]=rd(); bmp[t]=rd(); bsp[t]=rd(); emask[t]=rd(); }
        for (int t = 0; t < 3; ++t) { rd(); Ws[t]=rd(); rd(); bs[t]=rd(); maskp[t]=rd(); }
        x = (const int*)d_in[i++];
    }

    // ---- workspace carve (Bt buffers padded to multiples of 128 cols) ----
    char* base = (char*)d_ws;
    size_t off = 0;
    auto carve = [&](size_t bytes) -> void* {
        void* p = base + off;
        off += (bytes + 255) & ~(size_t)255;
        return p;
    };
    const int P_QKV  = npad128(3 * DIMC);     // 1536
    const int P_DIM  = npad128(DIMC);         // 512
    const int P_SEQ  = npad128(SEQD);         // 128
    const int P_MAIN = npad128(DIMC - SEQD);  // 512
    const int P_VOC  = npad128(VOCABC);       // 30080

    __bf16* wqkv_t[7]; __bf16* wp_t[7];
    for (int l = 0; l < 7; ++l) wqkv_t[l] = (__bf16*)carve((size_t)P_QKV * DIMC * 2);
    for (int l = 0; l < 7; ++l) wp_t[l]   = (__bf16*)carve((size_t)P_DIM * DIMC * 2);
    __bf16* wsp_t[2]; __bf16* wg_t[2]; __bf16* wmp_t[2]; __bf16* ws_t[3];
    for (int t = 0; t < 2; ++t) wsp_t[t] = (__bf16*)carve((size_t)P_SEQ * DIMC * 2);
    for (int t = 0; t < 2; ++t) wg_t[t]  = (__bf16*)carve((size_t)P_SEQ * DIMC * 2);
    for (int t = 0; t < 2; ++t) wmp_t[t] = (__bf16*)carve((size_t)P_MAIN * DIMC * 2);
    for (int t = 0; t < 3; ++t) ws_t[t]  = (__bf16*)carve((size_t)P_SEQ * DIMC * 2);
    __bf16* wout_t = (__bf16*)carve((size_t)P_VOC * DIMC * 2);
    __bf16* hb     = (__bf16*)carve((size_t)ROWS * DIMC * 2);
    __bf16* attb   = (__bf16*)carve((size_t)ROWS * DIMC * 2);
    float*  qkvf   = (float*)carve((size_t)ROWS * 3 * DIMC * 4);
    float*  gatef  = (float*)carve((size_t)ROWS * SEQD * 4);
    float*  seqf   = (float*)carve((size_t)ROWS * SEQD * 4);
    float*  mainf  = (float*)carve((size_t)ROWS * (DIMC - SEQD) * 4);

    // ---- weight conversions (fp32 -> transposed, zero-padded bf16) ----
    auto conv = [&](const float* W, __bf16* Wt, int N) {
        const int Np = npad128(N);
        const int total = Np * DIMC;
        convtrans_kernel<<<(total + 255) / 256, 256, 0, stream>>>(W, Wt, DIMC, N, Np);
    };
    for (int l = 0; l < 7; ++l) { conv(Wqkv[l], wqkv_t[l], 3 * DIMC); conv(Wp[l], wp_t[l], DIMC); }
    for (int t = 0; t < 2; ++t) { conv(Wsp[t], wsp_t[t], SEQD); conv(Wg[t], wg_t[t], SEQD);
                                  conv(Wmp[t], wmp_t[t], DIMC - SEQD); }
    for (int t = 0; t < 3; ++t) conv(Ws[t], ws_t[t], SEQD);
    conv(Wout, wout_t, VOCABC);

    auto gemm = [&](const __bf16* A, const __bf16* Bt, const float* bias,
                    const float* scale, float* Cf, __bf16* Cb, int N, int mode) {
        dim3 grid(npad128(N) / 128, ROWS / 64);
        if (mode == 1)
            gemm_bf16_kernel<1><<<grid, 256, 0, stream>>>(A, Bt, bias, scale, Cf, Cb, ROWS, N, DIMC);
        else
            gemm_bf16_kernel<0><<<grid, 256, 0, stream>>>(A, Bt, bias, scale, Cf, Cb, ROWS, N, DIMC);
    };

    // ---- pipeline ----
    float* out_f = (float*)d_out;
    float* seq0  = out_f + (size_t)ROWS * VOCABC;
    float* seq1  = seq0 + (size_t)ROWS * SEQD;
    float* seq2  = seq1 + (size_t)ROWS * SEQD;

    embed_kernel<<<(ROWS * DIMC + 255) / 256, 256, 0, stream>>>(x, emb, hb);

    gemm(hb, ws_t[0], bs[0], maskp[0], seq0, nullptr, SEQD, 0);

    auto attn_layer = [&](int li, int heads, int window) {
        const int hd = DIMC / heads;
        gemm(hb, wqkv_t[li], bqkv[li], nullptr, qkvf, nullptr, 3 * DIMC, 0);
        const float inv = 1.0f / sqrtf((float)hd);
        attn_kernel<<<BATCH * SEQLEN * heads, 256, 0, stream>>>(qkvf, attb, heads, hd, window, inv);
        gemm(attb, wp_t[li], bp[li], nullptr, nullptr, hb, DIMC, 0);
    };

    auto transition = [&](int t) {
        gemm(hb, wg_t[t],  bg[t],  nullptr, gatef, nullptr, SEQD, 1);
        gemm(hb, wsp_t[t], bsp[t], nullptr, seqf,  nullptr, SEQD, 0);
        gemm(hb, wmp_t[t], bmp[t], nullptr, mainf, nullptr, DIMC - SEQD, 0);
        transition_fuse_kernel<<<(ROWS * DIMC + 255) / 256, 256, 0, stream>>>(
            seqf, gatef, mainf, emask[t], hb);
    };

    for (int l = 0; l < 4; ++l) attn_layer(l, 8, 16);
    transition(0);
    gemm(hb, ws_t[1], bs[1], maskp[1], seq1, nullptr, SEQD, 0);
    for (int l = 4; l < 6; ++l) attn_layer(l, 4, 8);
    transition(1);
    gemm(hb, ws_t[2], bs[2], maskp[2], seq2, nullptr, SEQD, 0);
    attn_layer(6, 2, 4);

    gemm(hb, wout_t, bout, nullptr, out_f, nullptr, VOCABC, 0);
}
